// Base_55954833932808
// MI455X (gfx1250) — matrix operand order repaired and checked by reading, not compiled, not run on hardware
//
#include <hip/hip_runtime.h>
#include <stdint.h>

// ---------------- problem constants (match reference) ----------------
#define NND   16384      // total nodes N
#define HIDD  512        // hidden dim
#define EE    262144     // edges
#define BB    64         // batch (graphs)
#define NNODE 256        // nodes per graph
#define EPSV  1e-5f
#define OUTW  264        // GOUT(8) + NUM_NODES(256)

typedef __attribute__((ext_vector_type(16))) __bf16 v16bf;
typedef __attribute__((ext_vector_type(8)))  float  v8f;

__device__ __forceinline__ __bf16 f2bf(float f) {
  // round-to-nearest-even f32 -> bf16, bit-level
  uint32_t u = __builtin_bit_cast(uint32_t, f);
  uint32_t r = u + 0x7FFFu + ((u >> 16) & 1u);
  uint16_t h = (uint16_t)(r >> 16);
  return __builtin_bit_cast(__bf16, h);
}

// ---------------- elementwise helpers ----------------
__global__ void k_cvt_bf16(const float* __restrict__ in, __bf16* __restrict__ out, int n) {
  int i = blockIdx.x * blockDim.x + threadIdx.x;
  if (i < n) out[i] = f2bf(in[i]);
}

// grouped transposing convert: in f32 [G][R][C] -> out bf16 [G][C][R]
__global__ void k_cvtT_bf16(const float* __restrict__ in, __bf16* __restrict__ out,
                            int R, int C, int total) {
  int i = blockIdx.x * blockDim.x + threadIdx.x;
  if (i >= total) return;
  int rc = R * C;
  int g = i / rc;
  int rem = i - g * rc;
  int r = rem / C;
  int c = rem - r * C;
  out[(size_t)g * rc + (size_t)c * R + r] = f2bf(in[i]);
}

__global__ void k_zero(float* __restrict__ p, int n) {
  int i = blockIdx.x * blockDim.x + threadIdx.x;
  if (i < n) p[i] = 0.f;
}

// agg[dst[e]] += x[src[e]]  — one thread per (edge, channel)
__global__ void k_scatter_add(const float* __restrict__ x, const int* __restrict__ src,
                              const int* __restrict__ dst, float* __restrict__ agg) {
  long long i = (long long)blockIdx.x * blockDim.x + threadIdx.x;
  int e = (int)(i >> 9);
  int c = (int)(i & 511);
  float v = x[(size_t)src[e] * HIDD + c];
  unsafeAtomicAdd(&agg[(size_t)dst[e] * HIDD + c], v);
}

// per-column mean / biased var over all N rows (BatchNorm1d training)
__global__ void k_bn_stats(const float* __restrict__ h, float* __restrict__ mean,
                           float* __restrict__ var) {
  __shared__ float ss[256], sq[256];
  int c = blockIdx.x;
  int t = threadIdx.x;
  float s = 0.f, q = 0.f;
  for (int r = t; r < NND; r += 256) {
    float v = h[(size_t)r * HIDD + c];
    s += v; q += v * v;
  }
  ss[t] = s; sq[t] = q;
  __syncthreads();
  for (int w = 128; w > 0; w >>= 1) {
    if (t < w) { ss[t] += ss[t + w]; sq[t] += sq[t + w]; }
    __syncthreads();
  }
  if (t == 0) {
    float m = ss[0] / (float)NND;
    mean[c] = m;
    var[c]  = sq[0] / (float)NND - m * m;
  }
}

// x = relu(BN(h)); writes f32 + bf16 mirror
__global__ void k_bn_relu(const float* __restrict__ h, const float* __restrict__ mean,
                          const float* __restrict__ var, const float* __restrict__ gamma,
                          const float* __restrict__ beta, float* __restrict__ xf,
                          __bf16* __restrict__ xb) {
  int i = blockIdx.x * blockDim.x + threadIdx.x;
  int c = i & (HIDD - 1);
  float v = (h[i] - mean[c]) * rsqrtf(var[c] + EPSV) * gamma[c] + beta[c];
  v = fmaxf(v, 0.f);
  xf[i] = v;
  xb[i] = f2bf(v);
}

// graph mean-pool then ReLU (graph_shared starts with ReLU), bf16 out
__global__ void k_pool_relu(const float* __restrict__ xf, __bf16* __restrict__ xgb) {
  int i = blockIdx.x * blockDim.x + threadIdx.x;   // 64*512
  int g = i >> 9;
  int c = i & 511;
  const float* base = xf + ((size_t)g * NNODE) * HIDD + c;
  float s = 0.f;
  for (int n = 0; n < NNODE; ++n) s += base[(size_t)n * HIDD];
  float m = s * (1.f / (float)NNODE);
  xgb[i] = f2bf(fmaxf(m, 0.f));
}

// out_graph = g2 @ gh_w3 + gh_b3   ([64x256]·[256x8])
__global__ void k_graph_out(const float* __restrict__ g2, const float* __restrict__ w3,
                            const float* __restrict__ b3, float* __restrict__ out) {
  int i = blockIdx.x * blockDim.x + threadIdx.x;
  if (i >= BB * 8) return;
  int g = i >> 3, j = i & 7;
  const float* row = g2 + (size_t)g * 256;
  float s = b3[j];
  for (int k = 0; k < 256; ++k) s += row[k] * w3[k * 8 + j];
  out[(size_t)g * OUTW + j] = s;
}

// out_node[g][n] = dot(h2[n][g][:], nh_w3[n]) + nh_b3[n]
__global__ void k_node_out(const float* __restrict__ h2, const float* __restrict__ w3,
                           const float* __restrict__ b3, float* __restrict__ out) {
  int i = blockIdx.x * blockDim.x + threadIdx.x;   // 64*256
  int g = i >> 8, n = i & 255;
  const float* row = h2 + ((size_t)n * BB + g) * 128;
  const float* w   = w3 + (size_t)n * 128;
  float s = b3[n];
  for (int k = 0; k < 128; ++k) s += row[k] * w[k];
  out[(size_t)g * OUTW + 8 + n] = s;
}

// ---------------- WMMA fragment unions ----------------
union AFragU { v16bf v; uint32_t u[8]; };
union BFragU { v16bf v; uint32_t u[8]; };

// ---------------- generic WMMA GEMM (B pre-transposed: Bt[N][K]) ----------------
// Block = 128 threads (4 wave32), 64x64 tile; wave w -> rows [16w,16w+16).
// Grouped via blockIdx.z element strides. Bias / ReLU / optional bf16 mirror.
__global__ __launch_bounds__(128)
void k_gemm(const __bf16* __restrict__ A, long long lda, long long sAg,
            const __bf16* __restrict__ Bt, long long ldb /*==K*/, long long sBg,
            const float* __restrict__ bias, long long sBiasg,
            float* __restrict__ C, long long ldc, long long sCg,
            __bf16* __restrict__ Cbf, int K, int relu) {
  const int g = blockIdx.z;
  const __bf16* Ag   = A  + (size_t)g * sAg;
  const __bf16* Btg  = Bt + (size_t)g * sBg;
  float*        Cg   = C  + (size_t)g * sCg;
  __bf16*       Cbfg = Cbf ? (Cbf + (size_t)g * sCg) : nullptr;
  const float*  bg   = bias ? (bias + (size_t)g * sBiasg) : nullptr;

  const int lane = threadIdx.x & 31;
  const int wave = threadIdx.x >> 5;
  const int tileM   = blockIdx.y * 64 + wave * 16;
  const int colBase = blockIdx.x * 64;

  const int am  = lane & 15;
  const int akh = (lane >> 4) << 4;   // A: lanes 16-31 -> K rows +16
  const int bn  = lane & 15;
  const int bkh = (lane >> 4) << 4;   // B: lanes 16-31 -> K rows +16

  const __bf16* arow = Ag + (size_t)(tileM + am) * lda;
  const __bf16* brow0 = Btg + (size_t)(colBase + bn) * ldb + bkh;  // + 16j*ldb per tile

  v8f acc[4];
#pragma unroll
  for (int j = 0; j < 4; ++j) acc[j] = (v8f)0.0f;

  for (int k = 0; k < K; k += 32) {
    AFragU a;
#pragma unroll
    for (int v = 0; v < 8; ++v) {
      int k0 = k + akh + 2 * v;
      a.u[v] = *(const uint32_t*)(arow + k0);
    }
#pragma unroll
    for (int j = 0; j < 4; ++j) {
      BFragU b;
      const __bf16* br = brow0 + (size_t)(16 * j) * ldb + k;   // 32 contiguous bytes
#pragma unroll
      for (int v = 0; v < 8; ++v) b.u[v] = *(const uint32_t*)(br + 2 * v);
      acc[j] = __builtin_amdgcn_wmma_f32_16x16x32_bf16(
          false, a.v, false, b.v, (short)0, acc[j], false, false);
    }
  }

  const int crow0 = tileM + ((lane >> 4) << 3);
#pragma unroll
  for (int j = 0; j < 4; ++j) {
    int cc = colBase + 16 * j + (lane & 15);
    float bv = bg ? bg[cc] : 0.f;
#pragma unroll
    for (int r = 0; r < 8; ++r) {
      float val = acc[j][r] + bv;
      if (relu) val = fmaxf(val, 0.f);
      size_t off = (size_t)(crow0 + r) * ldc + cc;
      Cg[off] = val;
      if (Cbfg) Cbfg[off] = f2bf(val);
    }
  }
}

// ---------------- conv WMMA GEMM with async-LDS double-buffered B staging ----
// h = A1@B1 + A2@B2;  M=NND, N=K=HIDD.  B1t/B2t are [N][K] bf16.
// Per 32-K step each block stages both 64x32 B tiles (8 KB) into LDS with
// GLOBAL_LOAD_ASYNC_TO_LDS_B128 (ASYNCcnt), waits with s_wait_asynccnt,
// then all 4 waves read fragments from LDS (ds_load) — B shared, not 4x loaded.
__global__ __launch_bounds__(128)
void k_gemm_conv_async(const __bf16* __restrict__ A1, const __bf16* __restrict__ A2,
                       const __bf16* __restrict__ B1t, const __bf16* __restrict__ B2t,
                       float* __restrict__ C) {
  __shared__ __bf16 sB[2][2][64][32];   // [buf][mat][col][k]  = 16 KB

  const int tid  = threadIdx.x;
  const int lane = tid & 31;
  const int wave = tid >> 5;
  const int tileM   = blockIdx.y * 64 + wave * 16;
  const int colBase = blockIdx.x * 64;

  const int am  = lane & 15;
  const int akh = (lane >> 4) << 4;
  const int bn  = lane & 15;
  const int bkh = (lane >> 4) << 4;

  const __bf16* arow1 = A1 + (size_t)(tileM + am) * HIDD;
  const __bf16* arow2 = A2 + (size_t)(tileM + am) * HIDD;

  const int sc = tid >> 1;             // staging column 0..63
  const int s0 = (tid & 1) * 8;        // staging k-chunk base (8 bf16 = 16 B)

  v8f acc[4];
#pragma unroll
  for (int j = 0; j < 4; ++j) acc[j] = (v8f)0.0f;

  auto stage = [&](int buf, int kbase) {
#pragma unroll
    for (int h = 0; h < 2; ++h) {
      int koff = s0 + h * 16;          // {0,16} or {8,24}
      uint32_t l1 = (uint32_t)(uintptr_t)&sB[buf][0][sc][koff];
      unsigned long long g1 =
          (unsigned long long)(uintptr_t)(B1t + (size_t)(colBase + sc) * HIDD + kbase + koff);
      asm volatile("global_load_async_to_lds_b128 %0, %1, off"
                   :: "v"(l1), "v"(g1) : "memory");
      uint32_t l2 = (uint32_t)(uintptr_t)&sB[buf][1][sc][koff];
      unsigned long long g2 =
          (unsigned long long)(uintptr_t)(B2t + (size_t)(colBase + sc) * HIDD + kbase + koff);
      asm volatile("global_load_async_to_lds_b128 %0, %1, off"
                   :: "v"(l2), "v"(g2) : "memory");
    }
  };

  stage(0, 0);
  const int NK = HIDD / 32;            // 16 k-steps
  for (int kt = 0; kt < NK; ++kt) {
    asm volatile("s_wait_asynccnt 0x0" ::: "memory");
    __syncthreads();                   // staged tiles visible to all waves
    if (kt + 1 < NK) stage((kt + 1) & 1, (kt + 1) * 32);
    const int buf = kt & 1;
    const int k = kt * 32;

    AFragU a1, a2;
#pragma unroll
    for (int v = 0; v < 8; ++v) {
      int k0 = k + akh + 2 * v;
      a1.u[v] = *(const uint32_t*)(arow1 + k0);
      a2.u[v] = *(const uint32_t*)(arow2 + k0);
    }
#pragma unroll
    for (int j = 0; j < 4; ++j) {
      BFragU b1, b2;
      const __bf16* s1 = &sB[buf][0][16 * j + bn][bkh];
      const __bf16* s2 = &sB[buf][1][16 * j + bn][bkh];
#pragma unroll
      for (int v = 0; v < 8; ++v) {
        b1.u[v] = *(const uint32_t*)(s1 + 2 * v);
        b2.u[v] = *(const uint32_t*)(s2 + 2 * v);
      }
      acc[j] = __builtin_amdgcn_wmma_f32_16x16x32_bf16(
          false, a1.v, false, b1.v, (short)0, acc[j], false, false);
      acc[j] = __builtin_amdgcn_wmma_f32_16x16x32_bf16(
          false, a2.v, false, b2.v, (short)0, acc[j], false, false);
    }
    __syncthreads();                   // all waves done before buffer reuse
  }

  const int crow0 = tileM + ((lane >> 4) << 3);
#pragma unroll
  for (int j = 0; j < 4; ++j) {
    int cc = colBase + 16 * j + (lane & 15);
#pragma unroll
    for (int r = 0; r < 8; ++r)
      C[(size_t)(crow0 + r) * HIDD + cc] = acc[j][r];
  }
}

// ---------------- host orchestration ----------------
extern "C" void kernel_launch(void* const* d_in, const int* in_sizes, int n_in,
                              void* d_out, int out_size, void* d_ws, size_t ws_size,
                              hipStream_t stream) {
  (void)in_sizes; (void)n_in; (void)out_size; (void)ws_size;
  const float* x_in        = (const float*)d_in[0];
  const int*   ei          = (const int*)d_in[1];
  // d_in[2] batch: graph-major contiguous -> unused
  const float* conv_wself  = (const float*)d_in[3];
  const float* conv_wneigh = (const float*)d_in[4];
  // d_in[5] conv_b: per-column constant cancels exactly in (h - mean) -> skipped
  const float* bn_gamma = (const float*)d_in[6];
  const float* bn_beta  = (const float*)d_in[7];
  const float* gs_w1 = (const float*)d_in[8];  const float* gs_b1 = (const float*)d_in[9];
  const float* gs_w2 = (const float*)d_in[10]; const float* gs_b2 = (const float*)d_in[11];
  const float* gh_w1 = (const float*)d_in[12]; const float* gh_b1 = (const float*)d_in[13];
  const float* gh_w2 = (const float*)d_in[14]; const float* gh_b2 = (const float*)d_in[15];
  const float* gh_w3 = (const float*)d_in[16]; const float* gh_b3 = (const float*)d_in[17];
  const float* nh_w1 = (const float*)d_in[18]; const float* nh_b1 = (const float*)d_in[19];
  const float* nh_w2 = (const float*)d_in[20]; const float* nh_b2 = (const float*)d_in[21];
  const float* nh_w3 = (const float*)d_in[22]; const float* nh_b3 = (const float*)d_in[23];
  const int* src = ei;
  const int* dst = ei + EE;

  char* wsb = (char*)d_ws;
  size_t off = 0;
  auto alloc = [&](size_t bytes) -> void* {
    void* p = wsb + off;
    off += (bytes + 255) & ~(size_t)255;
    return p;
  };
  __bf16* xbf    = (__bf16*)alloc((size_t)NND * HIDD * 2);
  float*  aggf   = (float*) alloc((size_t)NND * HIDD * 4);
  __bf16* aggbf  = (__bf16*)alloc((size_t)NND * HIDD * 2);
  float*  hbuf   = aggf;   // safe alias: conv GEMM reads aggbf, writes hbuf
  float*  xf     = (float*) alloc((size_t)NND * HIDD * 4);
  float*  meanp  = (float*) alloc(HIDD * 4);
  float*  varp   = (float*) alloc(HIDD * 4);
  __bf16* wselfT  = (__bf16*)alloc((size_t)3 * HIDD * HIDD * 2);
  __bf16* wneighT = (__bf16*)alloc((size_t)3 * HIDD * HIDD * 2);
  __bf16* gsw1T = (__bf16*)alloc((size_t)HIDD * HIDD * 2);
  __bf16* gsw2T = (__bf16*)alloc((size_t)HIDD * HIDD * 2);
  __bf16* ghw1T = (__bf16*)alloc((size_t)HIDD * HIDD * 2);
  __bf16* ghw2T = (__bf16*)alloc((size_t)HIDD * 256 * 2);
  __bf16* nhw1T = (__bf16*)alloc((size_t)NNODE * HIDD * 256 * 2);
  __bf16* nhw2T = (__bf16*)alloc((size_t)NNODE * 256 * 128 * 2);
  __bf16* xgb = (__bf16*)alloc((size_t)BB * HIDD * 2);
  float*  t1f = (float*)alloc((size_t)BB * HIDD * 4);
  __bf16* t1b = (__bf16*)alloc((size_t)BB * HIDD * 2);
  float*  t2f = (float*)alloc((size_t)BB * HIDD * 4);
  __bf16* t2b = (__bf16*)alloc((size_t)BB * HIDD * 2);
  float*  g1f = (float*)alloc((size_t)BB * HIDD * 4);
  __bf16* g1b = (__bf16*)alloc((size_t)BB * HIDD * 2);
  float*  g2f = (float*)alloc((size_t)BB * 256 * 4);
  __bf16* g2b = (__bf16*)alloc((size_t)BB * 256 * 2);
  float*  h1f = (float*) alloc((size_t)NNODE * BB * 256 * 4);
  __bf16* h1b = (__bf16*)alloc((size_t)NNODE * BB * 256 * 2);
  float*  h2f = (float*) alloc((size_t)NNODE * BB * 128 * 4);

  const int T = 256;
  // activations: plain convert; all GEMM B operands: transposing convert
  k_cvt_bf16<<<(NND * HIDD) / T, T, 0, stream>>>(x_in, xbf, NND * HIDD);
  k_cvtT_bf16<<<(3 * HIDD * HIDD) / T, T, 0, stream>>>(conv_wself, wselfT, HIDD, HIDD, 3 * HIDD * HIDD);
  k_cvtT_bf16<<<(3 * HIDD * HIDD) / T, T, 0, stream>>>(conv_wneigh, wneighT, HIDD, HIDD, 3 * HIDD * HIDD);
  k_cvtT_bf16<<<(HIDD * HIDD) / T, T, 0, stream>>>(gs_w1, gsw1T, HIDD, HIDD, HIDD * HIDD);
  k_cvtT_bf16<<<(HIDD * HIDD) / T, T, 0, stream>>>(gs_w2, gsw2T, HIDD, HIDD, HIDD * HIDD);
  k_cvtT_bf16<<<(HIDD * HIDD) / T, T, 0, stream>>>(gh_w1, ghw1T, HIDD, HIDD, HIDD * HIDD);
  k_cvtT_bf16<<<(HIDD * 256) / T, T, 0, stream>>>(gh_w2, ghw2T, HIDD, 256, HIDD * 256);
  k_cvtT_bf16<<<(NNODE * HIDD * 256) / T, T, 0, stream>>>(nh_w1, nhw1T, HIDD, 256, NNODE * HIDD * 256);
  k_cvtT_bf16<<<(NNODE * 256 * 128) / T, T, 0, stream>>>(nh_w2, nhw2T, 256, 128, NNODE * 256 * 128);

  // ---- conv + BN + ReLU stack ----
  const float* xcur = x_in;
  for (int l = 0; l < 3; ++l) {
    k_zero<<<(NND * HIDD) / T, T, 0, stream>>>(aggf, NND * HIDD);
    k_scatter_add<<<(int)(((long long)EE * HIDD) / T), T, 0, stream>>>(xcur, src, dst, aggf);
    k_cvt_bf16<<<(NND * HIDD) / T, T, 0, stream>>>(aggf, aggbf, NND * HIDD);
    k_gemm_conv_async<<<dim3(HIDD / 64, NND / 64, 1), 128, 0, stream>>>(
        xbf, aggbf,
        wselfT + (size_t)l * HIDD * HIDD,
        wneighT + (size_t)l * HIDD * HIDD,
        hbuf);
    k_bn_stats<<<HIDD, 256, 0, stream>>>(hbuf, meanp, varp);
    k_bn_relu<<<(NND * HIDD) / T, T, 0, stream>>>(
        hbuf, meanp, varp, bn_gamma + (size_t)l * HIDD, bn_beta + (size_t)l * HIDD, xf, xbf);
    xcur = xf;
  }

  // ---- graph branch ----
  k_pool_relu<<<(BB * HIDD) / T, T, 0, stream>>>(xf, xgb);
  k_gemm<<<dim3(8, 1, 1), 128, 0, stream>>>(xgb, HIDD, 0, gsw1T, HIDD, 0,
      gs_b1, 0, t1f, HIDD, 0, t1b, HIDD, 0);
  k_gemm<<<dim3(8, 1, 1), 128, 0, stream>>>(t1b, HIDD, 0, gsw2T, HIDD, 0,
      gs_b2, 0, t2f, HIDD, 0, t2b, HIDD, 1);
  k_gemm<<<dim3(8, 1, 1), 128, 0, stream>>>(t2b, HIDD, 0, ghw1T, HIDD, 0,
      gh_b1, 0, g1f, HIDD, 0, g1b, HIDD, 1);
  k_gemm<<<dim3(4, 1, 1), 128, 0, stream>>>(g1b, HIDD, 0, ghw2T, HIDD, 0,
      gh_b2, 0, g2f, 256, 0, g2b, HIDD, 1);
  k_graph_out<<<2, 256, 0, stream>>>(g2f, gh_w3, gh_b3, (float*)d_out);

  // ---- node branch: 256-group strided GEMMs ----
  k_gemm<<<dim3(256 / 64, 1, NNODE), 128, 0, stream>>>(
      xbf, (long long)NNODE * HIDD, HIDD,
      nhw1T, HIDD, (long long)256 * HIDD,
      nh_b1, 256,
      h1f, 256, (long long)BB * 256, h1b, HIDD, 1);
  k_gemm<<<dim3(128 / 64, 1, NNODE), 128, 0, stream>>>(
      h1b, 256, (long long)BB * 256,
      nhw2T, 256, (long long)128 * 256,
      nh_b2, 128,
      h2f, 128, (long long)BB * 128, nullptr, 256, 1);
  k_node_out<<<BB, 256, 0, stream>>>(h2f, nh_w3, nh_b3, (float*)d_out);
}
